// GL_TAGConv_3l_512h_nw_k3_gnorm_relu_44753559224338
// MI455X (gfx1250) — matrix operand order repaired and checked by reading, not compiled, not run on hardware
//
#include <hip/hip_runtime.h>

// ---------------- problem constants (from reference) ----------------
#define NN      100000
#define FIN     128
#define HD      512
#define COUT    16
#define NE      3200000
#define GN_EPS  1e-5f

typedef __attribute__((ext_vector_type(16))) __bf16 v16bf;
typedef __attribute__((ext_vector_type(8)))  __bf16 v8bf;
typedef __attribute__((ext_vector_type(8)))  float  v8f;

__device__ __forceinline__ __bf16 f2bf(float f) {
    union { float f; unsigned int u; } x; x.f = f;
    unsigned int r = x.u + 0x7FFFu + ((x.u >> 16) & 1u);   // round-to-nearest-even
    union { unsigned short s; __bf16 b; } y; y.s = (unsigned short)(r >> 16);
    return y.b;
}

// ---------------- utility kernels ----------------
__global__ void zero_f32(float* __restrict__ p, size_t n) {
    size_t i = (size_t)blockIdx.x * blockDim.x + threadIdx.x;
    size_t st = (size_t)gridDim.x * blockDim.x;
    for (; i < n; i += st) p[i] = 0.0f;
}

__global__ void convert_bf16(const float* __restrict__ x, __bf16* __restrict__ y, size_t n) {
    size_t i = (size_t)blockIdx.x * blockDim.x + threadIdx.x;
    size_t st = (size_t)gridDim.x * blockDim.x;
    for (; i < n; i += st) y[i] = f2bf(x[i]);
}

// ---------------- gcn_norm ----------------
__global__ void deg_kernel(const int* __restrict__ ei, float* __restrict__ deg, int E) {
    int e = blockIdx.x * blockDim.x + threadIdx.x;
    if (e < E) atomicAdd(&deg[ei[(size_t)E + e]], 1.0f);   // col = target
}

__global__ void dinv_kernel(const float* __restrict__ deg, float* __restrict__ dinv, int n) {
    int i = blockIdx.x * blockDim.x + threadIdx.x;
    if (i < n) {
        float d = deg[i];
        dinv[i] = (d > 0.0f) ? rsqrtf(fmaxf(d, 1.0f)) : 0.0f;
    }
}

__global__ void edgenorm_kernel(const int* __restrict__ ei, const float* __restrict__ dinv,
                                float* __restrict__ nrm, int E) {
    int e = blockIdx.x * blockDim.x + threadIdx.x;
    if (e < E) nrm[e] = dinv[ei[e]] * dinv[ei[(size_t)E + e]];
}

// ---------------- sparse propagation: out[col] += nrm * h[row] ----------------
// one block per edge, blockDim = F/4, float4 gather + 4 float atomics scatter
__global__ void prop_kernel(const float* __restrict__ h, const float* __restrict__ nrm,
                            const int* __restrict__ ei, float* __restrict__ o,
                            int F, int E) {
    int e = blockIdx.x;
    int src = ei[e];
    int dst = ei[(size_t)E + e];
    float w = nrm[e];
    const float4* hv = (const float4*)(h + (size_t)src * F);
    float* od = o + (size_t)dst * F;
    float4 v = hv[threadIdx.x];
    int c = threadIdx.x * 4;
    atomicAdd(&od[c + 0], w * v.x);
    atomicAdd(&od[c + 1], w * v.y);
    atomicAdd(&od[c + 2], w * v.z);
    atomicAdd(&od[c + 3], w * v.w);
}

// ---------------- W fragment repack ----------------
// Layout: [hop k][kstep s][ntile][lane 0..31][e 0..15], B mapping ke = (e&7) + 8*(lane/16) + 16*(e>>3)
__global__ void pack_w(const float* __restrict__ W, __bf16* __restrict__ Wp,
                       int Fin, int Fout, int total) {
    int idx = blockIdx.x * blockDim.x + threadIdx.x;
    if (idx >= total) return;
    const int ntiles = Fout >> 4;
    const int ksteps = Fin >> 5;
    int e    = idx & 15;
    int lane = (idx >> 4) & 31;
    int rest = idx >> 9;
    int nt = rest % ntiles; rest /= ntiles;
    int s  = rest % ksteps;
    int k  = rest / ksteps;
    int gg = lane >> 4, nn = lane & 15;
    int kk = s * 32 + (e & 7) + 8 * gg + ((e >> 3) << 4);
    Wp[idx] = f2bf(W[(size_t)(k * Fin + kk) * Fout + nt * 16 + nn]);
}

// ---------------- full-width WMMA GEMM (Fout == 512, ntiles == 32) ----------------
// grid = (N/16), block = 256 (8 waves); each wave owns FOUR 16x16 output tiles
// (nbase = wave*4) -> one block covers the whole 512-wide output row band.
// Straight-line inner body: no guards, EXEC all-ones everywhere around WMMA.
__global__ __launch_bounds__(256)
void gemm_wmma_full(const __bf16* __restrict__ A, const __bf16* __restrict__ Wp,
                    float* __restrict__ out, int Fin, int accum) {
    __shared__ alignas(16) __bf16 atile[16 * 128];    // 16 rows x 128 k-cols (4 k-steps)
    const int mtile = blockIdx.x;
    const int wave  = threadIdx.x >> 5;
    const int lane  = threadIdx.x & 31;
    const int g  = lane >> 4;       // lane group
    const int ln = lane & 15;       // row (A) / col (B,D) within tile
    const int nbase = wave * 4;     // this wave's 4 output tiles (always valid: 32 total)
    v8f acc[4] = {};

    const int iters = Fin >> 7;     // 128 k per LDS stage
    const int t    = threadIdx.x;
    const int arow = t >> 4;
    const int acol = (t & 15) << 3; // 8 bf16 (16B) per thread
    const __bf16* asrc = A + (size_t)(mtile * 16 + arow) * Fin + acol;

    for (int it = 0; it < iters; ++it) {
        // cooperative A-tile stage into LDS (one b128 per thread)
        *(uint4*)&atile[arow * 128 + acol] = *(const uint4*)(asrc + it * 128);
        if (it + 1 < iters) __builtin_prefetch(asrc + (it + 1) * 128, 0, 1);
        __syncthreads();
#pragma unroll
        for (int s2 = 0; s2 < 4; ++s2) {
            const int s = it * 4 + s2;
            // A fragment: lane g*16+m holds ke = e(+8 if e>=8) + 8g -> two 16B chunks
            v8bf lo = *(const v8bf*)&atile[ln * 128 + s2 * 32 + 8 * g];
            v8bf hi = *(const v8bf*)&atile[ln * 128 + s2 * 32 + 16 + 8 * g];
            v16bf a = __builtin_shufflevector(lo, hi,
                      0,1,2,3,4,5,6,7,8,9,10,11,12,13,14,15);
            const __bf16* bbase = Wp + ((size_t)(s * 32 + nbase) * 32 + lane) * 16;
#pragma unroll
            for (int j = 0; j < 4; ++j) {
                const v16bf b = *(const v16bf*)(bbase + (size_t)j * 512);
                acc[j] = __builtin_amdgcn_wmma_f32_16x16x32_bf16(
                             false, a, false, b, (short)0, acc[j], false, false);
            }
        }
        __syncthreads();
    }
    // D layout: VGPR r -> row m = r + 8*g, col = ln
#pragma unroll
    for (int j = 0; j < 4; ++j) {
        float* o = out + (size_t)(mtile * 16 + 8 * g) * HD + (nbase + j) * 16 + ln;
        if (accum) {
#pragma unroll
            for (int r = 0; r < 8; ++r) o[(size_t)r * HD] += acc[j][r];
        } else {
#pragma unroll
            for (int r = 0; r < 8; ++r) o[(size_t)r * HD] = acc[j][r];
        }
    }
}

// ---------------- skinny WMMA GEMM (Fin == 512, Fout == 16, ntiles == 1) ----------
// one wave per block (blockDim=32), one 16x16 output tile per M-tile; A fragments
// loaded directly from global (no sharing to exploit), no barriers, no guards.
__global__ __launch_bounds__(32)
void gemm_wmma_skinny(const __bf16* __restrict__ A, const __bf16* __restrict__ Wp,
                      float* __restrict__ out, int accum) {
    const int mtile = blockIdx.x;
    const int lane  = threadIdx.x & 31;
    const int g  = lane >> 4;
    const int ln = lane & 15;
    v8f acc = {};
    const __bf16* arow = A + (size_t)(mtile * 16 + ln) * HD;   // lane's A row
#pragma unroll 4
    for (int s = 0; s < 16; ++s) {                              // 512 / 32 k-steps
        v8bf lo = *(const v8bf*)(arow + s * 32 + 8 * g);
        v8bf hi = *(const v8bf*)(arow + s * 32 + 16 + 8 * g);
        v16bf a = __builtin_shufflevector(lo, hi,
                  0,1,2,3,4,5,6,7,8,9,10,11,12,13,14,15);
        const v16bf b = *(const v16bf*)(Wp + ((size_t)s * 32 + lane) * 16);
        acc = __builtin_amdgcn_wmma_f32_16x16x32_bf16(
                  false, a, false, b, (short)0, acc, false, false);
    }
    float* o = out + (size_t)(mtile * 16 + 8 * g) * COUT + ln;
    if (accum) {
#pragma unroll
        for (int r = 0; r < 8; ++r) o[(size_t)r * COUT] += acc[r];
    } else {
#pragma unroll
        for (int r = 0; r < 8; ++r) o[(size_t)r * COUT] = acc[r];
    }
}

// ---------------- epilogues ----------------
__global__ void bias_act(float* __restrict__ x, const float* __restrict__ b,
                         int F, size_t n, int relu) {
    size_t i = (size_t)blockIdx.x * blockDim.x + threadIdx.x;
    size_t st = (size_t)gridDim.x * blockDim.x;
    for (; i < n; i += st) {
        float v = x[i] + b[i % F];
        if (relu) v = fmaxf(v, 0.0f);
        x[i] = v;
    }
}

// GraphNorm stats: per-feature sum and sum-of-squares (F fixed = 512)
__global__ void gn_stats(const float* __restrict__ x, float* __restrict__ st, int n) {
    int t = threadIdx.x;                 // 256 threads -> features t and t+256
    int r0 = blockIdx.x * 256;
    int r1 = min(r0 + 256, n);
    float s1a = 0, s2a = 0, s1b = 0, s2b = 0;
    for (int r = r0; r < r1; ++r) {
        float va = x[(size_t)r * 512 + t];
        float vb = x[(size_t)r * 512 + t + 256];
        s1a += va; s2a += va * va;
        s1b += vb; s2b += vb * vb;
    }
    atomicAdd(&st[t], s1a);         atomicAdd(&st[512 + t], s2a);
    atomicAdd(&st[t + 256], s1b);   atomicAdd(&st[512 + t + 256], s2b);
}

// y = x*scale + shift, scale = rsqrt(var+eps)*w, shift = b - a*m*scale
// var = q - (2a - a^2) m^2  with m = Sx/N, q = Sx2/N  (sub = x - a*m)
__global__ void gn_final(const float* __restrict__ st, const float* __restrict__ w,
                         const float* __restrict__ b, const float* __restrict__ a,
                         float* __restrict__ coef, int n) {
    int f = blockIdx.x * blockDim.x + threadIdx.x;
    if (f >= 512) return;
    float invn = 1.0f / (float)n;
    float m  = st[f] * invn;
    float q  = st[512 + f] * invn;
    float al = a[f];
    float var = q - (2.0f * al - al * al) * m * m;
    float s = rsqrtf(var + GN_EPS) * w[f];
    coef[f] = s;
    coef[512 + f] = b[f] - al * m * s;
}

__global__ void gn_apply(float* __restrict__ x, const float* __restrict__ coef, size_t n) {
    size_t i = (size_t)blockIdx.x * blockDim.x + threadIdx.x;
    size_t st = (size_t)gridDim.x * blockDim.x;
    for (; i < n; i += st) {
        int f = (int)(i & 511);
        x[i] = x[i] * coef[f] + coef[512 + f];
    }
}

// ---------------- host orchestration ----------------
extern "C" void kernel_launch(void* const* d_in, const int* in_sizes, int n_in,
                              void* d_out, int out_size, void* d_ws, size_t ws_size,
                              hipStream_t stream) {
    const float* x    = (const float*)d_in[0];
    const int*   ei   = (const int*)  d_in[1];
    const float* W1   = (const float*)d_in[2];
    const float* b1   = (const float*)d_in[3];
    const float* W2   = (const float*)d_in[4];
    const float* b2   = (const float*)d_in[5];
    const float* W3   = (const float*)d_in[6];
    const float* b3   = (const float*)d_in[7];
    const float* gn1w = (const float*)d_in[8];
    const float* gn1b = (const float*)d_in[9];
    const float* gn1a = (const float*)d_in[10];
    const float* gn2w = (const float*)d_in[11];
    const float* gn2b = (const float*)d_in[12];
    const float* gn2a = (const float*)d_in[13];
    float* out = (float*)d_out;

    char* ws = (char*)d_ws;
    size_t off = 0;
    auto alloc = [&](size_t bytes) -> void* {
        void* p = ws + off;
        off = (off + bytes + 255) & ~(size_t)255;
        return p;
    };
    float*  deg   = (float*) alloc((size_t)NN * 4);
    float*  dinv  = (float*) alloc((size_t)NN * 4);
    float*  nrm   = (float*) alloc((size_t)NE * 4);
    float*  bufA  = (float*) alloc((size_t)NN * HD * 4);   // prop ping
    float*  bufB  = (float*) alloc((size_t)NN * HD * 4);   // prop pong
    float*  bufC  = (float*) alloc((size_t)NN * HD * 4);   // layer1 out / layer2 in
    float*  bufD  = (float*) alloc((size_t)NN * HD * 4);   // layer2 out / layer3 in
    __bf16* Abf   = (__bf16*)alloc((size_t)NN * HD * 2);   // bf16 staging of current hop
    __bf16* Wp    = (__bf16*)alloc((size_t)4 * 16 * 32 * 512 * 2);  // packed W (max 2MB)
    float*  stats = (float*) alloc(1024 * 4);
    float*  coef  = (float*) alloc(1024 * 4);

    auto zero = [&](float* p, size_t n) {
        zero_f32<<<dim3(2048), 256, 0, stream>>>(p, n);
    };

    // ---- gcn_norm ----
    zero(deg, NN);
    deg_kernel     <<<(NE + 255) / 256, 256, 0, stream>>>(ei, deg, NE);
    dinv_kernel    <<<(NN + 255) / 256, 256, 0, stream>>>(deg, dinv, NN);
    edgenorm_kernel<<<(NE + 255) / 256, 256, 0, stream>>>(ei, dinv, nrm, NE);

    // ---- one TAGConv(+bias)(+ReLU)(+GraphNorm) layer ----
    auto run_layer = [&](const float* X, int Fin, int Fout,
                         const float* W, const float* bias, float* O, int relu,
                         const float* gw, const float* gb, const float* ga) {
        const int ntiles = Fout >> 4;
        const int ksteps = Fin >> 5;
        const int wtotal = 4 * ksteps * ntiles * 512;
        const size_t hopStride = (size_t)ksteps * ntiles * 512;
        const size_t nf = (size_t)NN * Fin;
        pack_w<<<(wtotal + 255) / 256, 256, 0, stream>>>(W, Wp, Fin, Fout, wtotal);

        auto gemm = [&](const __bf16* Ain, const __bf16* Wk, int accum) {
            if (Fout == COUT)
                gemm_wmma_skinny<<<NN / 16, 32, 0, stream>>>(Ain, Wk, O, accum);
            else
                gemm_wmma_full<<<NN / 16, 256, 0, stream>>>(Ain, Wk, O, Fin, accum);
        };

        // hop 0
        convert_bf16<<<2048, 256, 0, stream>>>(X, Abf, nf);
        gemm(Abf, Wp, 0);
        // hops 1..3
        const float* hsrc = X;
        float* hp[2] = {bufA, bufB};
        for (int k = 1; k <= 3; ++k) {
            float* dst = hp[(k - 1) & 1];   // k=1->A, k=2->B, k=3->A
            zero(dst, nf);
            prop_kernel<<<NE, Fin / 4, 0, stream>>>(hsrc, nrm, ei, dst, Fin, NE);
            convert_bf16<<<2048, 256, 0, stream>>>(dst, Abf, nf);
            gemm(Abf, Wp + k * hopStride, 1);
            hsrc = dst;
        }
        bias_act<<<2048, 256, 0, stream>>>(O, bias, Fout, (size_t)NN * Fout, relu);
        if (gw) {   // GraphNorm (Fout == 512 here)
            zero(stats, 1024);
            gn_stats<<<(NN + 255) / 256, 256, 0, stream>>>(O, stats, NN);
            gn_final<<<2, 256, 0, stream>>>(stats, gw, gb, ga, coef, NN);
            gn_apply<<<4096, 256, 0, stream>>>(O, coef, (size_t)NN * 512);
        }
    };

    run_layer(x,    FIN, HD,   W1, b1, bufC, 1, gn1w, gn1b, gn1a);
    run_layer(bufC, HD,  HD,   W2, b2, bufD, 1, gn2w, gn2b, gn2a);
    run_layer(bufD, HD,  COUT, W3, b3, out,  0, nullptr, nullptr, nullptr);
}
